// feature_conbined_regression_7533372637416
// MI455X (gfx1250) — matrix operand order repaired and checked by reading, not compiled, not run on hardware
//
#include <hip/hip_runtime.h>
#include <hip/hip_bf16.h>

typedef __attribute__((ext_vector_type(16))) _Float16 v16h;
typedef __attribute__((ext_vector_type(8)))  float    v8f;
typedef __attribute__((ext_vector_type(4)))  int      v4i;

#define AS1 __attribute__((address_space(1)))
#define AS3 __attribute__((address_space(3)))

#define BATCH 2048
#define XC    300
#define NWL   296
#define FM    74
#define HEAD  64
#define EMB   48
#define KTOT  (FM*FM)        // 5476
#define KT    172            // ceil(5476/32)
#define KPAD  (KT*32)        // 5504
#define NWAVE 8              // K-split waves per block

#if __has_builtin(__builtin_amdgcn_global_load_async_to_lds_b128) && \
    __has_builtin(__builtin_amdgcn_s_wait_asynccnt)
#define HAVE_ASYNC_LDS 1
#endif

// ---------------- workspace layout (bytes) ----------------
#define OFF_BPACK 0
#define OFF_PX    ((OFF_BPACK + 4*KT*32*16*2 + 255) & ~255)
#define OFF_VEFF  ((OFF_PX + BATCH*FM*4 + 255) & ~255)
#define OFF_HNDI  ((OFF_VEFF + HEAD*FM*4 + 255) & ~255)

__device__ __forceinline__ float elu_f(float v) {
    return v > 0.f ? v : (__expf(v) - 1.f);
}

// ---- pack wNDI into WMMA B-fragment order: [nt][t][lane][kk] ----
// B matrix 32x16 f16 layout: lanes 0-15 hold K=0..7,16..23, lanes 16-31 hold K=8..15,24..31,
// N = lane%16, per-lane 16 K values (2 per VGPR).
__global__ void k_bpack(const float* __restrict__ wNDI, _Float16* __restrict__ Bpack) {
    int idx = blockIdx.x * blockDim.x + threadIdx.x;
    if (idx >= 4 * KT * 32 * 16) return;
    int kk   = idx & 15;
    int lane = (idx >> 4) & 31;
    int t    = (idx >> 9) % KT;
    int nt   = idx / (KT * 512);
    int h    = nt * 16 + (lane & 15);
    int k    = t * 32 + ((lane < 16) ? 0 : 8) + (kk < 8 ? kk : kk + 8);
    float v  = (k < KTOT) ? wNDI[(size_t)h * KTOT + k] : 0.f;
    Bpack[idx] = (_Float16)v;
}

// ---- collapsed DI weight: Veff[h][i] = colsum_j wDI[h][j][i] - rowsum_j wDI[h][i][j] ----
__global__ void k_veff(const float* __restrict__ wDI, float* __restrict__ Veff) {
    int idx = blockIdx.x * blockDim.x + threadIdx.x;
    if (idx >= HEAD * FM) return;
    int h = idx / FM, i = idx % FM;
    const float* w = wDI + (size_t)h * KTOT;
    float cs = 0.f, rs = 0.f;
    for (int j = 0; j < FM; ++j) {
        cs += w[j * FM + i];
        rs += w[i * FM + j];
    }
    Veff[idx] = cs - rs;
}

// ---- pooled x: px[b][i] = mean of 4 ----
__global__ void k_px(const float* __restrict__ x, float* __restrict__ px) {
    int idx = blockIdx.x * blockDim.x + threadIdx.x;
    if (idx >= BATCH * FM) return;
    int b = idx / FM, i = idx % FM;
    const float* xr = x + (size_t)b * XC + i * 4;
    px[idx] = 0.25f * (xr[0] + xr[1] + xr[2] + xr[3]);
}

// ---- fused NDI-pool + GEMM: h_NDI[b][h] = sum_k pNDI(b,k) * wNDI(h,k) ----
// 128 blocks, one 16-batch M-tile each; 8 waves split the K-tiles; WMMA f16->f32.
__global__ __launch_bounds__(32 * NWAVE) void k_gemm(const float* __restrict__ x,
                                                     const _Float16* __restrict__ Bpack,
                                                     float* __restrict__ hNDI) {
    __shared__ float xs[16 * NWL];
    __shared__ float cacc[16 * HEAD];
    const int tid   = threadIdx.x;
    const int lane  = tid & 31;
    const int wave  = tid >> 5;
    const int mbase = blockIdx.x * 16;

    // ---- stage 16 x-rows into LDS (async-to-LDS engine when available) ----
#ifdef HAVE_ASYNC_LDS
    for (int v = tid; v < 16 * (NWL / 4); v += 32 * NWAVE) {   // 16B granules
        int bb = v / (NWL / 4), c4 = v % (NWL / 4);
        __builtin_amdgcn_global_load_async_to_lds_b128(
            (AS1 v4i*)(x + (size_t)(mbase + bb) * XC + c4 * 4),
            (AS3 v4i*)&xs[bb * NWL + c4 * 4],
            /*offset=*/0, /*cpol=*/0);
    }
    __builtin_amdgcn_s_wait_asynccnt(0);
#else
    for (int i = tid; i < 16 * NWL; i += 32 * NWAVE) {
        int bb = i / NWL, col = i % NWL;
        xs[i] = x[(size_t)(mbase + bb) * XC + col];
    }
#endif
    for (int i = tid; i < 16 * HEAD; i += 32 * NWAVE) cacc[i] = 0.f;
    __syncthreads();

    const int m = lane & 15;
    const float* xrow = &xs[m * NWL];
    const int kbase = (lane < 16) ? 0 : 8;   // A-fragment K split per ISA table
    const size_t ntstride = (size_t)KT * 32 * 16;

    v8f c0 = {}; v8f c1 = {}; v8f c2 = {}; v8f c3 = {};

    for (int t = wave; t < KT; t += NWAVE) {
        // ---- synthesize A fragment: pNDI(b, k) on the fly, f16, branchless pad ----
        v16h a;
#pragma unroll
        for (int kk = 0; kk < 16; ++kk) {
            int k   = t * 32 + kbase + (kk < 8 ? kk : kk + 8);
            int inb = (k < KTOT);
            int kc  = inb ? k : 0;
            int I = kc / FM, J = kc % FM;
            const float* xa = xrow + J * 4;
            const float* xb = xrow + I * 4;
            float s = 0.f;
#pragma unroll
            for (int q = 0; q < 4; ++q) {
                float aq = xa[q];
#pragma unroll
                for (int p = 0; p < 4; ++p) {
                    float bp  = xb[p];
                    float num = aq - bp;
                    float den = aq + bp + 1e-5f;
                    s += num * __builtin_amdgcn_rcpf(den);
                }
            }
            a[kk] = (_Float16)(inb ? s * 0.0625f : 0.f);
        }

        // ---- load 4 B fragments (L2-resident, reused by all blocks) ----
        const _Float16* bp = Bpack + ((size_t)t * 32 + lane) * 16;
        __builtin_prefetch(bp + NWAVE * 32 * 16, 0, 0);   // next K-tile for this wave
        v16h b0 = *(const v16h*)(bp);
        v16h b1 = *(const v16h*)(bp + ntstride);
        v16h b2 = *(const v16h*)(bp + 2 * ntstride);
        v16h b3 = *(const v16h*)(bp + 3 * ntstride);

        c0 = __builtin_amdgcn_wmma_f32_16x16x32_f16(false, a, false, b0, (short)0, c0, false, false);
        c1 = __builtin_amdgcn_wmma_f32_16x16x32_f16(false, a, false, b1, (short)0, c1, false, false);
        c2 = __builtin_amdgcn_wmma_f32_16x16x32_f16(false, a, false, b2, (short)0, c2, false, false);
        c3 = __builtin_amdgcn_wmma_f32_16x16x32_f16(false, a, false, b3, (short)0, c3, false, false);
    }

    // ---- combine K-split partials across the waves via LDS float atomics ----
    // C layout: vgpr r -> M = r + (lane<16?0:8), N = nt*16 + (lane&15)
    const int mr = (lane >> 4) * 8;
    const int hc = lane & 15;
#pragma unroll
    for (int r = 0; r < 8; ++r) {
        atomicAdd(&cacc[(r + mr) * HEAD +  0 + hc], c0[r]);
        atomicAdd(&cacc[(r + mr) * HEAD + 16 + hc], c1[r]);
        atomicAdd(&cacc[(r + mr) * HEAD + 32 + hc], c2[r]);
        atomicAdd(&cacc[(r + mr) * HEAD + 48 + hc], c3[r]);
    }
    __syncthreads();

    for (int i = tid; i < 16 * HEAD; i += 32 * NWAVE)
        hNDI[(size_t)(mbase + i / HEAD) * HEAD + (i % HEAD)] = cacc[i];
}

// ---- per-batch tail: DI branch (collapsed), bias+ELU, fc layers, sigmoid ----
__global__ __launch_bounds__(128) void k_tail(
    const float* __restrict__ px, const float* __restrict__ Veff,
    const float* __restrict__ hNDI,
    const float* __restrict__ bDI,  const float* __restrict__ fcDIw,  const float* __restrict__ fcDIb,
    const float* __restrict__ bNDI, const float* __restrict__ fcNDIw, const float* __restrict__ fcNDIb,
    const float* __restrict__ fc1w, const float* __restrict__ fc1b,
    const float* __restrict__ l1w,  const float* __restrict__ l1b,
    const float* __restrict__ l2w,  const float* __restrict__ l2b,
    float* __restrict__ out) {
    int b = blockIdx.x * blockDim.x + threadIdx.x;
    if (b >= BATCH) return;

    float e1[EMB];
    {
        float pxl[FM];
        for (int i = 0; i < FM; ++i) pxl[i] = px[(size_t)b * FM + i];
        float hv[HEAD];
        for (int h = 0; h < HEAD; ++h) {
            float s = bDI[h];
            const float* v = Veff + h * FM;
            for (int i = 0; i < FM; ++i) s += pxl[i] * v[i];
            hv[h] = elu_f(s);
        }
        for (int e = 0; e < EMB; ++e) {
            float s = fcDIb[e];
            const float* w = fcDIw + e * HEAD;
            for (int h = 0; h < HEAD; ++h) s += w[h] * hv[h];
            e1[e] = elu_f(s);
        }
    }
    float e2[EMB];
    {
        float hv[HEAD];
        for (int h = 0; h < HEAD; ++h) hv[h] = elu_f(hNDI[(size_t)b * HEAD + h] + bNDI[h]);
        for (int e = 0; e < EMB; ++e) {
            float s = fcNDIb[e];
            const float* w = fcNDIw + e * HEAD;
            for (int h = 0; h < HEAD; ++h) s += w[h] * hv[h];
            e2[e] = elu_f(s);
        }
    }
    float f1[EMB];
    for (int o = 0; o < EMB; ++o) {
        float s = fc1b[o];
        const float* w = fc1w + o * (2 * EMB);
        for (int i = 0; i < EMB; ++i) s += w[i] * e1[i];
        for (int i = 0; i < EMB; ++i) s += w[EMB + i] * e2[i];
        f1[o] = elu_f(s);
    }
    float g[20];
    for (int o = 0; o < 20; ++o) {
        float s = l1b[o];
        const float* w = l1w + o * EMB;
        for (int i = 0; i < EMB; ++i) s += w[i] * f1[i];
        g[o] = elu_f(s);
    }
    float z = l2b[0];
    for (int i = 0; i < 20; ++i) z += l2w[i] * g[i];
    out[b] = 1.f / (1.f + __expf(-z));
}

extern "C" void kernel_launch(void* const* d_in, const int* in_sizes, int n_in,
                              void* d_out, int out_size, void* d_ws, size_t ws_size,
                              hipStream_t stream) {
    const float* x      = (const float*)d_in[0];
    const float* wDI    = (const float*)d_in[1];
    const float* bDI    = (const float*)d_in[2];
    const float* fcDIw  = (const float*)d_in[3];
    const float* fcDIb  = (const float*)d_in[4];
    const float* wNDI   = (const float*)d_in[5];
    const float* bNDI   = (const float*)d_in[6];
    const float* fcNDIw = (const float*)d_in[7];
    const float* fcNDIb = (const float*)d_in[8];
    const float* fc1w   = (const float*)d_in[9];
    const float* fc1b   = (const float*)d_in[10];
    const float* l1w    = (const float*)d_in[11];
    const float* l1b    = (const float*)d_in[12];
    const float* l2w    = (const float*)d_in[13];
    const float* l2b    = (const float*)d_in[14];
    float* out = (float*)d_out;

    char* ws = (char*)d_ws;
    _Float16* Bpack = (_Float16*)(ws + OFF_BPACK);
    float*    px    = (float*)(ws + OFF_PX);
    float*    Veff  = (float*)(ws + OFF_VEFF);
    float*    hNDI  = (float*)(ws + OFF_HNDI);

    k_bpack<<<(4 * KT * 32 * 16 + 255) / 256, 256, 0, stream>>>(wNDI, Bpack);
    k_veff <<<(HEAD * FM + 255) / 256,        256, 0, stream>>>(wDI, Veff);
    k_px   <<<(BATCH * FM + 255) / 256,       256, 0, stream>>>(x, px);

    k_gemm <<<BATCH / 16, 32 * NWAVE, 0, stream>>>(x, Bpack, hNDI);

    k_tail <<<(BATCH + 127) / 128, 128, 0, stream>>>(px, Veff, hNDI,
        bDI, fcDIw, fcDIb, bNDI, fcNDIw, fcNDIb,
        fc1w, fc1b, l1w, l1b, l2w, l2b, out);
}
